// MultiheadAttention_86380382257374
// MI455X (gfx1250) — compile-verified, matrix operand order checked by reading, not run on hardware
//
#include <hip/hip_runtime.h>
#include <hip/hip_bf16.h>

// ---------------------------------------------------------------------------
// MultiheadAttention (quirky reference) on MI455X / gfx1250.
//
// Reference collapses algebraically to: out = 2048*((X@Wv + bv) @ Wo) + bo
// (softmax rows each sum to 1; the einsum sums weights over both q and k).
//
// fp32 inputs are split into bf16 hi/lo pairs; each GEMM is computed as
// hi*hi + hi*lo + lo*hi with v_wmma_f32_16x16x32_bf16 (f32 accumulate).
//
// Data movement: GLOBAL_LOAD_ASYNC_TO_LDS_B128 (ASYNCcnt-tracked) into a
// 3-deep LDS ring buffer, one barrier per K-step, one tile always in flight.
// Falls back to synchronous VGPR-staged copies if the builtin is absent.
// ---------------------------------------------------------------------------

typedef __attribute__((ext_vector_type(16))) __bf16 v16bf;
typedef __attribute__((ext_vector_type(8)))  float  v8f;
typedef int v4i __attribute__((vector_size(4 * sizeof(int))));

#define AS1 __attribute__((address_space(1)))
#define AS3 __attribute__((address_space(3)))

#if defined(__HIP_DEVICE_COMPILE__) && __has_builtin(__builtin_amdgcn_global_load_async_to_lds_b128)
#define MHA_ASYNC 1
#else
#define MHA_ASYNC 0
#endif

// ---- GEMM tiling constants -------------------------------------------------
constexpr int BM = 128, BN = 64, BK = 32;
constexpr int LSTR = 56;                       // padded LDS row stride (elems), 112 B
constexpr int A_OFF_H = 0;
constexpr int A_OFF_L = BM * LSTR;             // 7168
constexpr int B_OFF_H = 2 * BM * LSTR;         // 14336
constexpr int B_OFF_L = 2 * BM * LSTR + BN * LSTR;
constexpr int BUF_ELEMS = 2 * (BM + BN) * LSTR; // 21504 elems = 43008 B / buffer

union FragB {
    uint4 q[2];
    v16bf v;
};

__device__ __forceinline__ void split_f32(float x, unsigned short& h, unsigned short& l) {
    __bf16 bh = (__bf16)x;             // RNE convert
    float  fh = (float)bh;
    __bf16 bl = (__bf16)(x - fh);      // residual
    h = __builtin_bit_cast(unsigned short, bh);
    l = __builtin_bit_cast(unsigned short, bl);
}

#if MHA_ASYNC
// One 16-byte async global->LDS copy per lane (ASYNCcnt-tracked).
__device__ __forceinline__ void async_ld16(const unsigned short* g, unsigned short* l) {
    // generic global ptr value == AS1 address; generic LDS ptr low 32 bits == LDS offset
    __builtin_amdgcn_global_load_async_to_lds_b128(
        (AS1 v4i*)(uintptr_t)g,
        (AS3 v4i*)(unsigned int)(uintptr_t)l,
        0, 0);
}

template <int N>
__device__ __forceinline__ void wait_asynccnt_le() {
#if __has_builtin(__builtin_amdgcn_s_wait_asynccnt)
    __builtin_amdgcn_s_wait_asynccnt(N);
#else
    asm volatile("s_wait_asynccnt %0" :: "i"(N) : "memory");
#endif
}

// Raw workgroup barrier without the compiler fence (which could force a full
// asynccnt drain); compiler-level memory barriers block code motion, and the
// wave's ds_loads issue strictly after s_barrier_wait retires.
__device__ __forceinline__ void block_barrier() {
    asm volatile("" ::: "memory");
    __builtin_amdgcn_s_barrier();
    asm volatile("" ::: "memory");
}

// Issue all global->LDS async copies for one K-tile: 6 instructions per wave.
__device__ __forceinline__ void issue_tile_async(
    const unsigned short* __restrict__ Ah, const unsigned short* __restrict__ Al,
    const unsigned short* __restrict__ Bth, const unsigned short* __restrict__ Btl,
    unsigned short* buf, int tid, int mBlock, int nBlock, int K, int k0)
{
#pragma unroll
    for (int it = 0; it < 2; ++it) {
        int seg = tid + it * 256;
        int r = seg >> 2;
        int s = (seg & 3) << 3;
        size_t g = (size_t)(mBlock + r) * K + (k0 + s);
        int d = r * LSTR + s;
        async_ld16(Ah + g, buf + A_OFF_H + d);
        async_ld16(Al + g, buf + A_OFF_L + d);
    }
    {
        int r = tid >> 2;
        int s = (tid & 3) << 3;
        size_t g = (size_t)(nBlock + r) * K + (k0 + s);
        int d = r * LSTR + s;
        async_ld16(Bth + g, buf + B_OFF_H + d);
        async_ld16(Btl + g, buf + B_OFF_L + d);
    }
}
#endif // MHA_ASYNC

// 12 WMMAs for one K-step of the wave's 32x32 tile (3xBF16 compensation).
__device__ __forceinline__ void gemm_tile_compute(
    const unsigned short* buf, int waveM, int waveN, int e0, int frow, v8f (&acc)[2][2])
{
    FragB aH[2], aL[2], bH[2], bL[2];
#pragma unroll
    for (int mt = 0; mt < 2; ++mt) {
        int base = (waveM + mt * 16 + frow) * LSTR + e0;
        aH[mt].q[0] = *(const uint4*)(buf + A_OFF_H + base);
        aH[mt].q[1] = *(const uint4*)(buf + A_OFF_H + base + 16);
        aL[mt].q[0] = *(const uint4*)(buf + A_OFF_L + base);
        aL[mt].q[1] = *(const uint4*)(buf + A_OFF_L + base + 16);
    }
#pragma unroll
    for (int nt = 0; nt < 2; ++nt) {
        int base = (waveN + nt * 16 + frow) * LSTR + e0;
        bH[nt].q[0] = *(const uint4*)(buf + B_OFF_H + base);
        bH[nt].q[1] = *(const uint4*)(buf + B_OFF_H + base + 16);
        bL[nt].q[0] = *(const uint4*)(buf + B_OFF_L + base);
        bL[nt].q[1] = *(const uint4*)(buf + B_OFF_L + base + 16);
    }
#pragma unroll
    for (int mt = 0; mt < 2; ++mt)
#pragma unroll
        for (int nt = 0; nt < 2; ++nt) {
            acc[mt][nt] = __builtin_amdgcn_wmma_f32_16x16x32_bf16(
                false, aH[mt].v, false, bH[nt].v, (short)0, acc[mt][nt], false, false);
            acc[mt][nt] = __builtin_amdgcn_wmma_f32_16x16x32_bf16(
                false, aH[mt].v, false, bL[nt].v, (short)0, acc[mt][nt], false, false);
            acc[mt][nt] = __builtin_amdgcn_wmma_f32_16x16x32_bf16(
                false, aL[mt].v, false, bH[nt].v, (short)0, acc[mt][nt], false, false);
        }
}

// ---------------------------------------------------------------------------
// Prep: elementwise split fp32 -> (bf16 hi, bf16 lo)
// ---------------------------------------------------------------------------
__global__ __launch_bounds__(256) void split_bf16_kernel(
    const float* __restrict__ x,
    unsigned short* __restrict__ hi,
    unsigned short* __restrict__ lo,
    int n)
{
    int i = blockIdx.x * blockDim.x + threadIdx.x;
    if (i < n) {
        unsigned short h, l;
        split_f32(x[i], h, l);
        hi[i] = h;
        lo[i] = l;
    }
}

// ---------------------------------------------------------------------------
// Prep: transpose W[K][N] -> T[N][K] while splitting into bf16 hi/lo.
// Block (32,8), grid (N/32, K/32).
// ---------------------------------------------------------------------------
__global__ __launch_bounds__(256) void transpose_split_kernel(
    const float* __restrict__ W,
    unsigned short* __restrict__ Th,
    unsigned short* __restrict__ Tl,
    int K, int N)
{
    __shared__ float tile[32][33];
    const int n0 = blockIdx.x * 32;
    const int k0 = blockIdx.y * 32;
    const int tx = threadIdx.x;
    const int ty = threadIdx.y;

#pragma unroll
    for (int j = 0; j < 32; j += 8)
        tile[ty + j][tx] = W[(size_t)(k0 + ty + j) * N + (n0 + tx)];

    __syncthreads();

#pragma unroll
    for (int j = 0; j < 32; j += 8) {
        float v = tile[tx][ty + j];
        unsigned short h, l;
        split_f32(v, h, l);
        size_t idx = (size_t)(n0 + ty + j) * K + (k0 + tx);
        Th[idx] = h;
        Tl[idx] = l;
    }
}

// ---------------------------------------------------------------------------
// 3xBF16 WMMA GEMM:  C[M][N] = scale * (A[M][K] @ B[K][N] + bias[N])
// A: bf16 hi/lo row-major [M][K];  B: pre-transposed bf16 hi/lo [N][K].
// MODE 0: write C as bf16 hi/lo split.  MODE 1: write C as fp32.
// 256 threads = 8 waves; block tile 128x64; wave tile 32x32; BK = 32.
// Dynamic LDS: 3 ring buffers (async path) / 1 used (fallback path).
// ---------------------------------------------------------------------------
template <int MODE>
__global__ __launch_bounds__(256) void mha_gemm_3xbf16(
    const unsigned short* __restrict__ Ah,
    const unsigned short* __restrict__ Al,
    const unsigned short* __restrict__ Bth,
    const unsigned short* __restrict__ Btl,
    const float* __restrict__ bias,
    float scale,
    unsigned short* __restrict__ Ch,
    unsigned short* __restrict__ Cl,
    float* __restrict__ Cf,
    int M, int N, int K)
{
    (void)M;
    extern __shared__ unsigned short smem[];

    const int tid   = threadIdx.x;
    const int lane  = tid & 31;
    const int wave  = tid >> 5;
    const int waveM = (wave & 3) * 32;  // 4 waves along M
    const int waveN = (wave >> 2) * 32; // 2 waves along N
    const int mBlock = blockIdx.y * BM;
    const int nBlock = blockIdx.x * BN;

    v8f acc[2][2];
#pragma unroll
    for (int i = 0; i < 2; ++i)
#pragma unroll
        for (int j = 0; j < 2; ++j)
#pragma unroll
            for (int e = 0; e < 8; ++e)
                acc[i][j][e] = 0.0f;

    // CDNA5 16-bit A-matrix fragment addressing: lane l holds row (l&15);
    // K elements {e0..e0+7} and {e0+16..e0+23}, e0 = (l>=16) ? 8 : 0.
    const int e0   = (lane >> 4) << 3;
    const int frow = lane & 15;

#if MHA_ASYNC
    // ---- async pipeline: 3-deep LDS ring, one tile always in flight ----
    // Ring-buffer bases computed arithmetically (no pointer array: a constant
    // array of generic pointers to LDS is not linkable).
    const int S = K / BK;

    issue_tile_async(Ah, Al, Bth, Btl, smem, tid, mBlock, nBlock, K, 0);
    issue_tile_async(Ah, Al, Bth, Btl, smem + BUF_ELEMS, tid, mBlock, nBlock, K, BK);

    int ib = 0;
    for (int i = 0; i < S; ++i) {
        // 6 async loads per tile per wave -> <=6 pending means tile i landed.
        if (i + 1 < S) wait_asynccnt_le<6>();
        else           wait_asynccnt_le<0>();
        block_barrier();   // all waves' tile-i data visible in LDS

        if (i + 2 < S) {
            int inext = ib + 2; if (inext >= 3) inext -= 3;
            issue_tile_async(Ah, Al, Bth, Btl, smem + inext * BUF_ELEMS,
                             tid, mBlock, nBlock, K, (i + 2) * BK);
        }
        gemm_tile_compute(smem + ib * BUF_ELEMS, waveM, waveN, e0, frow, acc);
        if (++ib == 3) ib = 0;
    }
#else
    // ---- fallback: synchronous VGPR-staged copies, single buffer ----
    for (int k0 = 0; k0 < K; k0 += BK) {
#pragma unroll
        for (int it = 0; it < 2; ++it) {
            int seg = tid + it * 256;
            int r = seg >> 2;
            int s = (seg & 3) << 3;
            size_t g = (size_t)(mBlock + r) * K + (k0 + s);
            *(uint4*)(smem + A_OFF_H + r * LSTR + s) = *(const uint4*)(Ah + g);
            *(uint4*)(smem + A_OFF_L + r * LSTR + s) = *(const uint4*)(Al + g);
        }
        {
            int r = tid >> 2;
            int s = (tid & 3) << 3;
            size_t g = (size_t)(nBlock + r) * K + (k0 + s);
            *(uint4*)(smem + B_OFF_H + r * LSTR + s) = *(const uint4*)(Bth + g);
            *(uint4*)(smem + B_OFF_L + r * LSTR + s) = *(const uint4*)(Btl + g);
        }
        __syncthreads();
        gemm_tile_compute(smem, waveM, waveN, e0, frow, acc);
        __syncthreads();
    }
#endif

    // ---- epilogue ----
    // C/D layout: VGPR i, lane l -> row = i + ((l>=16)?8:0), col = l&15.
#pragma unroll
    for (int mt = 0; mt < 2; ++mt) {
#pragma unroll
        for (int nt = 0; nt < 2; ++nt) {
            int col = nBlock + waveN + nt * 16 + frow;
            float bcol = bias[col];
            int rbase = mBlock + waveM + mt * 16 + ((lane >> 4) << 3);
#pragma unroll
            for (int i = 0; i < 8; ++i) {
                float val = (acc[mt][nt][i] + bcol) * scale;
                size_t idx = (size_t)(rbase + i) * N + col;
                if (MODE == 0) {
                    unsigned short h, l;
                    split_f32(val, h, l);
                    Ch[idx] = h;
                    Cl[idx] = l;
                } else {
                    Cf[idx] = val;
                }
            }
        }
    }
}

// ---------------------------------------------------------------------------
// Launch
// ---------------------------------------------------------------------------
extern "C" void kernel_launch(void* const* d_in, const int* in_sizes, int n_in,
                              void* d_out, int out_size, void* d_ws, size_t ws_size,
                              hipStream_t stream)
{
    (void)in_sizes; (void)n_in; (void)out_size; (void)ws_size;

    // setup_inputs order:
    // 0 input_data [2,2048,512] f32   1 mask (unused)
    // 2 Wq (unused) 3 bq (unused) 4 Wk (unused) 5 bk (unused)
    // 6 Wv [512,4096] f32  7 bv [4096] f32
    // 8 Wo [4096,512] f32  9 bo [512] f32
    const float* X  = (const float*)d_in[0];
    const float* Wv = (const float*)d_in[6];
    const float* bv = (const float*)d_in[7];
    const float* Wo = (const float*)d_in[8];
    const float* bo = (const float*)d_in[9];
    float* out = (float*)d_out;

    const int Mrows = 2 * 2048;  // batch*seq = 4096
    const int D     = 512;       // d_model
    const int H     = 4096;      // num_heads * d_model

    const size_t szXD = (size_t)Mrows * D; // 2,097,152
    const size_t szDH = (size_t)D * H;     // 2,097,152
    const size_t szMH = (size_t)Mrows * H; // 16,777,216

    char* p = (char*)d_ws;
    unsigned short* Xh    = (unsigned short*)p; p += szXD * 2;
    unsigned short* Xl    = (unsigned short*)p; p += szXD * 2;
    unsigned short* Wvth  = (unsigned short*)p; p += szDH * 2;  // [H][D]
    unsigned short* Wvtl  = (unsigned short*)p; p += szDH * 2;
    unsigned short* Woth  = (unsigned short*)p; p += szDH * 2;  // [D][H]
    unsigned short* Wotl  = (unsigned short*)p; p += szDH * 2;
    unsigned short* Vsh   = (unsigned short*)p; p += szMH * 2;  // [Mrows][H]
    unsigned short* Vsl   = (unsigned short*)p; p += szMH * 2;

    const unsigned smemBytes = 3u * BUF_ELEMS * sizeof(unsigned short); // 129,024 B

    // Split X -> bf16 hi/lo
    split_bf16_kernel<<<dim3((unsigned)(szXD / 256)), dim3(256), 0, stream>>>(
        X, Xh, Xl, (int)szXD);

    // Wv [D][H] -> Wvt [H][D] hi/lo
    transpose_split_kernel<<<dim3(H / 32, D / 32), dim3(32, 8), 0, stream>>>(
        Wv, Wvth, Wvtl, D, H);

    // Wo [H][D] -> Wot [D][H] hi/lo
    transpose_split_kernel<<<dim3(D / 32, H / 32), dim3(32, 8), 0, stream>>>(
        Wo, Woth, Wotl, H, D);

    // GEMM1: Vs = 2048 * (X @ Wv + bv)   -> bf16 hi/lo split [Mrows][H]
    mha_gemm_3xbf16<0><<<dim3(H / 64, Mrows / 128), dim3(256), smemBytes, stream>>>(
        Xh, Xl, Wvth, Wvtl, bv, 2048.0f,
        Vsh, Vsl, nullptr, Mrows, H, D);

    // GEMM2: out = Vs @ Wo + bo          -> fp32 [Mrows][D]
    mha_gemm_3xbf16<1><<<dim3(D / 64, Mrows / 128), dim3(256), smemBytes, stream>>>(
        Vsh, Vsl, Woth, Wotl, bo, 1.0f,
        nullptr, nullptr, out, Mrows, D, H);
}
